// VectorQuantizer_26293789786234
// MI455X (gfx1250) — matrix operand order repaired and checked by reading, not compiled, not run on hardware
//
#include <hip/hip_runtime.h>
#include <math.h>

// ---------------------------------------------------------------------------
// VQ-VAE vector quantizer for MI455X (gfx1250, wave32, WMMA bf16 + TDM)
//   inputs [16,256,32,32] f32, emb [8192,256] f32
//   out = [loss(1)] [quantized 16*256*32*32] [perplexity(1)] [encodings 16384*8192]
// ---------------------------------------------------------------------------

typedef __attribute__((ext_vector_type(16))) __bf16 v16bf;
typedef __attribute__((ext_vector_type(8)))  __bf16 v8bf;
typedef __attribute__((ext_vector_type(8)))  float  v8f;
typedef __attribute__((ext_vector_type(4)))  unsigned int u32x4;
typedef __attribute__((ext_vector_type(8)))  int          i32x8;
typedef __attribute__((ext_vector_type(4)))  int          i32x4;

#define NUM_EMB   8192
#define DIM       256
#define NTOK      16384            // 16 * 32 * 32
#define QOUT_N    ((size_t)NTOK * DIM)
#define ENC_OFF   (1 + 4194304 + 1)   // loss, quantized, perplexity, then encodings

// LDS row stride for staged B tiles: 512B data + 16B TDM pad = 528B (264 bf16)
#define BROW      264

// ---- TDM: 1-D 8KB tile (16 codebook rows) global -> LDS, with LDS padding --
__device__ __forceinline__ void tdm_load_tile(unsigned ldsByteAddr,
                                              const __bf16* gsrc) {
  unsigned long long ga = (unsigned long long)(uintptr_t)gsrc;
  u32x4 g0;
  g0[0] = 1u;                                   // count=1, user descriptor
  g0[1] = ldsByteAddr;                          // LDS destination (bytes)
  g0[2] = (unsigned)ga;                         // global_addr[31:0]
  g0[3] = (unsigned)((ga >> 32) & 0x1FFFFFFu) | 0x80000000u;  // addr[56:32] | type=2

  i32x8 g1;
  // data_size=3 (8B) | pad_enable | pad_interval=6 (128 DW) | pad_amount=3 (4 DW)
  g1[0] = (3 << 16) | (1 << 20) | (6 << 22) | (3 << 25);
  g1[1] = (int)(1024u << 16);                   // tensor_dim0[15:0]=1024 in [31:16]
  g1[2] = (int)(1u << 16);                      // tensor_dim0 hi=0 | tensor_dim1=1
  g1[3] = (int)(1024u << 16);                   // tensor_dim1 hi=0 | tile_dim0=1024
  g1[4] = 0;                                    // tile_dim1=0, tile_dim2=0 (1-D)
  g1[5] = 1024;                                 // tensor_dim0_stride (unused, benign)
  g1[6] = 0;
  g1[7] = 0;

  i32x4 z4 = {0, 0, 0, 0};
#if __clang_major__ >= 23
  i32x8 z8 = {0, 0, 0, 0, 0, 0, 0, 0};
  __builtin_amdgcn_tensor_load_to_lds(g0, g1, z4, z4, z8, 0);
#else
  __builtin_amdgcn_tensor_load_to_lds(g0, g1, z4, z4, 0);
#endif
}

// ---- 1) codebook -> bf16, plus ||e||^2 ------------------------------------
__global__ __launch_bounds__(256) void k_emb_prep(const float* __restrict__ emb,
                                                  __bf16* __restrict__ embB,
                                                  float* __restrict__ eNorm) {
  __shared__ float red[256];
  const int row = blockIdx.x;
  const int c   = threadIdx.x;
  float v = emb[(size_t)row * DIM + c];
  embB[(size_t)row * DIM + c] = (__bf16)v;
  red[c] = v * v;
  __syncthreads();
  for (int s = 128; s > 0; s >>= 1) { if (c < s) red[c] += red[c + s]; __syncthreads(); }
  if (c == 0) eNorm[row] = red[0];
}

// ---- 2) BCHW -> [N, C] transpose, f32 -> bf16 (LDS tiled) ------------------
__global__ __launch_bounds__(256) void k_flatten(const float* __restrict__ in,
                                                 __bf16* __restrict__ flatB) {
  __shared__ float tile[32][33];
  const int b   = blockIdx.z;
  const int c0  = blockIdx.y * 32;
  const int hw0 = blockIdx.x * 32;
  const int tx = threadIdx.x, ty = threadIdx.y;   // 32 x 8
#pragma unroll
  for (int r = 0; r < 4; ++r) {
    int c = c0 + ty + r * 8;
    tile[ty + r * 8][tx] = in[((size_t)(b * DIM + c)) * 1024 + hw0 + tx];
  }
  __syncthreads();
#pragma unroll
  for (int r = 0; r < 4; ++r) {
    int n = b * 1024 + hw0 + ty + r * 8;
    flatB[(size_t)n * DIM + c0 + tx] = (__bf16)tile[tx][ty + r * 8];
  }
}

// ---- 3) fused bf16-WMMA GEMM + row argmin, TDM-staged B --------------------
// 4 waves / block, each wave owns a 16-row tile (64 rows per block). Codebook
// tiles (16 rows x 256 bf16 = 8KB) are DMA'd into LDS by the Tensor Data
// Mover, double buffered; all 4 waves consume the shared tile.
__global__ __launch_bounds__(128) void k_argmin(const __bf16* __restrict__ flatB,
                                                const __bf16* __restrict__ embB,
                                                const float* __restrict__ eNorm,
                                                int* __restrict__ outIdx) {
  __shared__ __bf16 sB[2][16 * BROW];   // 2 x 8448 B, TDM-padded rows

  const int tid  = threadIdx.x;
  const int wave = tid >> 5;             // 0..3
  const int l    = tid & 31;
  const int m    = l & 15;               // row (A) / col (B) within tile
  const int hi   = (l >> 4) & 1;         // lane half selects K sub-range
  const int rowBase = blockIdx.x * 64 + wave * 16;

  // Preload A: lane l<16 needs K 32k+0..15 (contiguous); lanes 16-31 hold +16.
  const __bf16* arow = flatB + (size_t)(rowBase + m) * DIM;
  v16bf A[8];
#pragma unroll
  for (int k = 0; k < 8; ++k) {
    v8bf lo = *(const v8bf*)(arow + k * 32 + hi * 16);
    v8bf hh = *(const v8bf*)(arow + k * 32 + hi * 16 + 8);
    A[k] = __builtin_shufflevector(lo, hh, 0,1,2,3,4,5,6,7,8,9,10,11,12,13,14,15);
  }

  float runMin[8]; int runIdx[8];
#pragma unroll
  for (int v = 0; v < 8; ++v) { runMin[v] = __builtin_huge_valf(); runIdx[v] = 0; }

  const unsigned ldsBase0 = (unsigned)(uintptr_t)&sB[0][0];
  const unsigned ldsBase1 = (unsigned)(uintptr_t)&sB[1][0];

  // Prologue: stage tile 0, wait, publish.
  if (wave == 0) {
    tdm_load_tile(ldsBase0, embB);
    __builtin_amdgcn_s_wait_tensorcnt(0);
  }
  __syncthreads();

  for (int j = 0; j < NUM_EMB / 16; ++j) {
    const int cur = j & 1;
    if (wave == 0 && j + 1 < NUM_EMB / 16) {
      tdm_load_tile(cur ? ldsBase0 : ldsBase1,
                    embB + (size_t)(j + 1) * 16 * DIM);
    }

    const int e = j * 16 + m;                       // this lane's candidate column
    const __bf16* brow = &sB[cur][m * BROW];
    const float en = eNorm[e];

    v8f acc0 = {}; v8f acc1 = {};
#pragma unroll
    for (int k = 0; k < 8; k += 2) {
      // B lane l<16 holds K 32k..32k+15 (contiguous); lanes 16-31 hold +16.
      v8bf b0a = *(const v8bf*)(brow + k * 32 + hi * 16);
      v8bf b0b = *(const v8bf*)(brow + k * 32 + hi * 16 + 8);
      v8bf b1a = *(const v8bf*)(brow + (k + 1) * 32 + hi * 16);
      v8bf b1b = *(const v8bf*)(brow + (k + 1) * 32 + hi * 16 + 8);
      v16bf B0 = __builtin_shufflevector(b0a, b0b, 0,1,2,3,4,5,6,7,8,9,10,11,12,13,14,15);
      v16bf B1 = __builtin_shufflevector(b1a, b1b, 0,1,2,3,4,5,6,7,8,9,10,11,12,13,14,15);
      acc0 = __builtin_amdgcn_wmma_f32_16x16x32_bf16(false, A[k],     false, B0,
                                                     (short)0, acc0, false, false);
      acc1 = __builtin_amdgcn_wmma_f32_16x16x32_bf16(false, A[k + 1], false, B1,
                                                     (short)0, acc1, false, false);
    }
    v8f acc = acc0 + acc1;

#pragma unroll
    for (int v = 0; v < 8; ++v) {                   // VGPR v -> row v (lanes<16) / v+8
      float d = fmaf(-2.0f, acc[v], en);
      bool better = d < runMin[v];                  // strict '<' keeps first occurrence
      runMin[v] = better ? d : runMin[v];
      runIdx[v] = better ? e : runIdx[v];
    }

    // Make tile j+1 resident before anyone flips buffers; barrier also keeps
    // the TDM for j+2 from overwriting a buffer still being read.
    if (wave == 0 && j + 1 < NUM_EMB / 16) {
      __builtin_amdgcn_s_wait_tensorcnt(0);
    }
    __syncthreads();
  }

  // Butterfly argmin across the 16 lanes of each half (tie -> smaller index).
#pragma unroll
  for (int v = 0; v < 8; ++v) {
#pragma unroll
    for (int off = 1; off < 16; off <<= 1) {
      float om = __shfl_xor(runMin[v], off, 32);
      int   oi = __shfl_xor(runIdx[v], off, 32);
      if (om < runMin[v] || (om == runMin[v] && oi < runIdx[v])) {
        runMin[v] = om; runIdx[v] = oi;
      }
    }
  }
  if (m == 0) {                                     // lanes 0 and 16
#pragma unroll
    for (int v = 0; v < 8; ++v) outIdx[rowBase + hi * 8 + v] = runIdx[v];
  }
}

// ---- 4) zero the one-hot region (8B aligned offset -> b64 stores) ----------
__global__ void k_zero64(float2* __restrict__ p, size_t count) {
  size_t i = (size_t)blockIdx.x * blockDim.x + threadIdx.x;
  const size_t stride = (size_t)gridDim.x * blockDim.x;
  const float2 z = {0.f, 0.f};
  for (; i < count; i += stride) p[i] = z;
}

__global__ void k_scatter(const int* __restrict__ idx, float* __restrict__ enc) {
  int n = blockIdx.x * blockDim.x + threadIdx.x;
  if (n < NTOK) enc[(size_t)n * NUM_EMB + idx[n]] = 1.0f;
}

__global__ void k_hist_zero(int* __restrict__ h) {
  int k = blockIdx.x * blockDim.x + threadIdx.x;
  if (k < NUM_EMB) h[k] = 0;
}

__global__ void k_hist(const int* __restrict__ idx, int* __restrict__ h) {
  int n = blockIdx.x * blockDim.x + threadIdx.x;
  if (n < NTOK) atomicAdd(&h[idx[n]], 1);
}

// ---- 5) gather quantized (BCHW layout) + per-block loss partials -----------
__global__ __launch_bounds__(256) void k_gather_loss(const float* __restrict__ in,
                                                     const float* __restrict__ emb,
                                                     const int* __restrict__ idx,
                                                     float* __restrict__ qout,
                                                     float* __restrict__ partial) {
  __shared__ float red[256];
  const int b  = blockIdx.z;
  const int hw = blockIdx.x * 32 + threadIdx.x;
  const int c  = blockIdx.y * 8 + threadIdx.y;
  const int n  = b * 1024 + hw;
  const int e  = idx[n];
  float q = emb[(size_t)e * DIM + c];
  size_t off = ((size_t)(b * DIM + c)) * 1024 + hw;
  float x = in[off];
  qout[off] = q;                                    // straight-through fwd == quantized
  float d = q - x;
  int t = threadIdx.y * 32 + threadIdx.x;
  red[t] = d * d;
  __syncthreads();
  for (int s = 128; s > 0; s >>= 1) { if (t < s) red[t] += red[t + s]; __syncthreads(); }
  if (t == 0)
    partial[((size_t)blockIdx.z * gridDim.y + blockIdx.y) * gridDim.x + blockIdx.x] = red[0];
}

// ---- 6) deterministic fixed-order finals -----------------------------------
__global__ __launch_bounds__(256) void k_loss_final(const float* __restrict__ partial,
                                                    float* __restrict__ out) {
  __shared__ float red[256];
  int t = threadIdx.x; float s = 0.f;
  for (int i = t; i < NTOK; i += 256) s += partial[i];
  red[t] = s; __syncthreads();
  for (int k = 128; k > 0; k >>= 1) { if (t < k) red[t] += red[t + k]; __syncthreads(); }
  // q_latent_loss == e_latent_loss numerically -> loss = 1.25 * MSE
  if (t == 0) out[0] = 1.25f * red[0] / (float)QOUT_N;
}

__global__ __launch_bounds__(256) void k_perplexity(const int* __restrict__ h,
                                                    float* __restrict__ out) {
  __shared__ float red[256];
  int t = threadIdx.x; float s = 0.f;
  for (int k = t; k < NUM_EMB; k += 256) {
    float p = (float)h[k] * (1.0f / (float)NTOK);
    s += p * logf(p + 1e-10f);
  }
  red[t] = s; __syncthreads();
  for (int k = 128; k > 0; k >>= 1) { if (t < k) red[t] += red[t + k]; __syncthreads(); }
  if (t == 0) out[0] = expf(-red[0]);
}

// ---------------------------------------------------------------------------
extern "C" void kernel_launch(void* const* d_in, const int* in_sizes, int n_in,
                              void* d_out, int out_size, void* d_ws, size_t ws_size,
                              hipStream_t stream) {
  const float* inputs = (const float*)d_in[0];   // [16,256,32,32]
  const float* emb    = (const float*)d_in[1];   // [8192,256]
  float* out = (float*)d_out;

  // workspace layout (~12.2 MB)
  char* ws = (char*)d_ws;
  __bf16* embB  = (__bf16*)(ws);                              // 4 MB
  __bf16* flatB = (__bf16*)(ws + (size_t)4  * 1024 * 1024);   // 8 MB
  float*  eNorm = (float*) (ws + (size_t)12 * 1024 * 1024);   // 32 KB
  int*    idx   = (int*)   (ws + (size_t)12 * 1024 * 1024 + 32  * 1024); // 64 KB
  int*    hist  = (int*)   (ws + (size_t)12 * 1024 * 1024 + 96  * 1024); // 32 KB
  float*  part  = (float*) (ws + (size_t)12 * 1024 * 1024 + 128 * 1024); // 64 KB

  float* qout    = out + 1;
  float* perpOut = out + 1 + QOUT_N;
  float* enc     = out + ENC_OFF;

  k_emb_prep<<<NUM_EMB, 256, 0, stream>>>(emb, embB, eNorm);
  k_flatten<<<dim3(32, 8, 16), dim3(32, 8), 0, stream>>>(inputs, flatB);
  k_argmin<<<NTOK / 64, 128, 0, stream>>>(flatB, embB, eNorm, idx);

  const size_t enc2 = (size_t)NTOK * NUM_EMB / 2;             // float2 count
  k_zero64<<<8192, 256, 0, stream>>>((float2*)enc, enc2);
  k_scatter<<<NTOK / 256, 256, 0, stream>>>(idx, enc);

  k_hist_zero<<<NUM_EMB / 256, 256, 0, stream>>>(hist);
  k_hist<<<NTOK / 256, 256, 0, stream>>>(idx, hist);

  k_gather_loss<<<dim3(32, 32, 16), dim3(32, 8), 0, stream>>>(inputs, emb, idx, qout, part);
  k_loss_final<<<1, 256, 0, stream>>>(part, out);
  k_perplexity<<<1, 256, 0, stream>>>(hist, perpOut);
}
